// LlamaAttention_420906795803
// MI455X (gfx1250) — matrix operand order repaired and checked by reading, not compiled, not run on hardware
//
#include <hip/hip_runtime.h>

// ---------------------------------------------------------------------------
// LLaMA attention block for MI455X (gfx1250, wave32).
//   B=2 S=2048 D=4096 H=32 KV=8 HD=128
// GEMMs on v_wmma_f32_16x16x32_bf16 (fp32 accum); weight tiles double-buffered
// in LDS via global_load_async_to_lds_b128 (ASYNCcnt); flash attention with
// shfl-based online softmax and an LDS C->A transpose for P. Hidden states
// and weights are pre-converted to bf16 once so every hot loop moves bf16.
// ---------------------------------------------------------------------------

typedef __attribute__((ext_vector_type(16))) __bf16 v16bf;
typedef __attribute__((ext_vector_type(8)))  __bf16 v8bf;
typedef __attribute__((ext_vector_type(8)))  float  v8f;

#define DI __device__ __forceinline__

constexpr int Bc = 2, Sc = 2048, Dc = 4096, Hc = 32, KVc = 8, HDc = 128;
constexpr float SCALE  = 0.08838834764831845f;   // 1/sqrt(128)
constexpr float NEGBIG = -1.0e30f;               // causal mask value
constexpr float ROPE_C = 0.2076205069f;          // log2(10000)/64

DI v8f wmma_bf16(v16bf a, v16bf b, v8f c) {
  // D = A(16x32 bf16) * B(32x16 bf16) + C(16x16 f32)
  return __builtin_amdgcn_wmma_f32_16x16x32_bf16(false, a, false, b,
                                                 (short)0, c, false, false);
}

// Build a 16x32 A-operand for one lane from two contiguous 8-elem chunks:
// lane(half) holds K in [half*16, half*16+16), contiguous (p1 = p0 + 8).
DI v16bf ldA(const __bf16* p0, const __bf16* p1) {
  v8bf x = *(const v8bf*)p0;
  v8bf y = *(const v8bf*)p1;
  v16bf r;
#pragma unroll
  for (int i = 0; i < 8; ++i) { r[i] = x[i]; r[8 + i] = y[i]; }
  return r;
}

// Async-copy one 128x32 bf16 tile (8KB) from global into LDS.
// 256 threads x 2 chunks x 16B = 8KB; 2 wave-instructions -> ASYNCcnt += 2.
DI void prefetch_wtile(const __bf16* Wb, int wrow, int kb, __bf16* tile,
                       int tid) {
  const unsigned long long base = (unsigned long long)(size_t)Wb;
#pragma unroll
  for (int rep = 0; rep < 2; ++rep) {
    const int ci = tid + rep * 256;          // 16-byte chunk index (0..511)
    const int o = ci >> 2;                   // row 0..127
    const int dl = (ci & 3) * 8;             // col 0,8,16,24
    const unsigned lds = (unsigned)(size_t)(tile + o * 32 + dl);
    const unsigned goff =
        (unsigned)(((size_t)(wrow + o) * Dc + kb + dl) * sizeof(__bf16));
    asm volatile("global_load_async_to_lds_b128 %0, %1, %2"
                 :: "v"(lds), "v"(goff), "s"(base) : "memory");
  }
}

// ---------------------------------------------------------------------------
// Kernel 0: fp32 -> bf16 elementwise convert (weights + hidden states).
// ---------------------------------------------------------------------------
__global__ __launch_bounds__(256) void cvt_bf16(const float* __restrict__ src,
                                                __bf16* __restrict__ dst,
                                                int n) {
  const int i = (blockIdx.x * 256 + threadIdx.x) * 8;
  if (i >= n) return;
  float4 a = *(const float4*)(src + i);
  float4 b = *(const float4*)(src + i + 4);
  v8bf o;
  o[0]=(__bf16)a.x; o[1]=(__bf16)a.y; o[2]=(__bf16)a.z; o[3]=(__bf16)a.w;
  o[4]=(__bf16)b.x; o[5]=(__bf16)b.y; o[6]=(__bf16)b.z; o[7]=(__bf16)b.w;
  *(v8bf*)(dst + i) = o;
}

// ---------------------------------------------------------------------------
// Kernel 1: fused QKV projection + RoPE.
//   grid = (B*S/128, 48)   block = 256 (8 waves)
//   job 0..31  -> Q head (RoPE, store [b][h][s][hd] bf16)
//   job 32..39 -> K head (RoPE, store [b][kv][s][hd] bf16)
//   job 40..47 -> V head (store TRANSPOSED [b][kv][hd][s] bf16)
// Each wave computes a 16(row) x 128(full head) strip so the RoPE partner
// (col +/- 64) sits in accumulator slice sl +/- 4 of the same lane.
// Weight k-chunks are double-buffered in LDS via async-to-LDS copies.
// ---------------------------------------------------------------------------
__global__ __launch_bounds__(256) void qkv_rope(
    const __bf16* __restrict__ hsb, const __bf16* __restrict__ Wqb,
    const __bf16* __restrict__ Wkb, const __bf16* __restrict__ Wvb,
    __bf16* __restrict__ qws, __bf16* __restrict__ kws,
    __bf16* __restrict__ vtws) {
  __shared__ __attribute__((aligned(64))) __bf16 wtile[2][128 * 32];

  const int job = blockIdx.y;
  const __bf16* W; int wrow, mode, head;
  if (job < Hc)            { W = Wqb; wrow = job * HDc;              mode = 0; head = job; }
  else if (job < Hc + KVc) { W = Wkb; wrow = (job - Hc) * HDc;       mode = 1; head = job - Hc; }
  else                     { W = Wvb; wrow = (job - Hc - KVc) * HDc; mode = 2; head = job - Hc - KVc; }

  const int tid = threadIdx.x, wave = tid >> 5, lane = tid & 31;
  const int n = lane & 15, half = lane >> 4;
  const int rowbase = blockIdx.x * 128;
  const __bf16* hrow = hsb + (size_t)(rowbase + wave * 16 + n) * Dc;

  v8f acc[8];
#pragma unroll
  for (int i = 0; i < 8; ++i)
#pragma unroll
    for (int j = 0; j < 8; ++j) acc[i][j] = 0.f;

  prefetch_wtile(W, wrow, 0, &wtile[0][0], tid);

  int it = 0;
  for (int kb = 0; kb < Dc; kb += 32, ++it) {
    const int nxt = kb + 32;
    if (nxt < Dc) {
      prefetch_wtile(W, wrow, nxt, &wtile[(it + 1) & 1][0], tid);
      asm volatile("s_wait_asynccnt 0x2" ::: "memory"); // current tile done
    } else {
      asm volatile("s_wait_asynccnt 0x0" ::: "memory");
    }
    __syncthreads();               // all waves' copies for this tile landed

    const __bf16* tile = &wtile[it & 1][0];
    v16bf a = ldA(hrow + kb + half * 16, hrow + kb + half * 16 + 8);
#pragma unroll
    for (int sl = 0; sl < 8; ++sl) {
      v16bf b = *(const v16bf*)&tile[(sl * 16 + n) * 32 + half * 16];
      acc[sl] = wmma_bf16(a, b, acc[sl]);
    }
    __syncthreads();               // done reading before it's overwritten
  }

  const int bIdx = rowbase / Sc;
  const int sbase = (rowbase % Sc) + wave * 16;
#pragma unroll
  for (int sl = 0; sl < 8; ++sl) {
#pragma unroll
    for (int v = 0; v < 8; ++v) {
      const int r = v + 8 * half;
      const int s = sbase + r;
      const int col = sl * 16 + n;
      const float val = acc[sl][v];
      if (mode == 2) {
        vtws[((size_t)(bIdx * KVc + head) * HDc + col) * Sc + s] = (__bf16)val;
      } else {
        const float partner = (sl < 4) ? acc[sl + 4][v] : acc[sl - 4][v];
        const float rot = (sl < 4) ? -partner : partner;
        const float theta = (float)s * exp2f(-(float)(col & 63) * ROPE_C);
        float sn, cs; __sincosf(theta, &sn, &cs);
        const float o = val * cs + rot * sn;
        if (mode == 0)
          qws[((size_t)(bIdx * Hc + head) * Sc + s) * HDc + col] = (__bf16)o;
        else
          kws[((size_t)(bIdx * KVc + head) * Sc + s) * HDc + col] = (__bf16)o;
      }
    }
  }
}

// ---------------------------------------------------------------------------
// Kernel 2: causal flash attention (GQA).  One wave per 16-row Q strip.
//   grid = B*H*(S/16)/8 blocks of 8 waves.
// ---------------------------------------------------------------------------
__global__ __launch_bounds__(256) void attn(
    const __bf16* __restrict__ qws, const __bf16* __restrict__ kws,
    const __bf16* __restrict__ vtws, __bf16* __restrict__ aws) {
  __shared__ __attribute__((aligned(64))) __bf16 pbuf[8][16 * 32];

  const int tid = threadIdx.x, wave = tid >> 5, lane = tid & 31;
  const int n = lane & 15, half = lane >> 4;
  const int jid = blockIdx.x * 8 + wave;
  const int b   = jid / (Hc * (Sc / 16));
  const int rem = jid % (Hc * (Sc / 16));
  const int h   = rem / (Sc / 16);
  const int qs  = (rem % (Sc / 16)) * 16;
  const int kv  = h >> 2;                       // n_rep = 4

  const __bf16* qrow = qws + ((size_t)(b * Hc + h) * Sc + qs + n) * HDc;
  v16bf aq[4];
#pragma unroll
  for (int c = 0; c < 4; ++c)
    aq[c] = ldA(qrow + c * 32 + half * 16, qrow + c * 32 + half * 16 + 8);

  v8f o[8];
  float mx[8], ls[8];
#pragma unroll
  for (int sl = 0; sl < 8; ++sl) {
#pragma unroll
    for (int j = 0; j < 8; ++j) o[sl][j] = 0.f;
    mx[sl] = -3.0e38f; ls[sl] = 0.f;
  }

  const __bf16* kbase = kws  + (size_t)(b * KVc + kv) * Sc * HDc;
  const __bf16* vbase = vtws + (size_t)(b * KVc + kv) * HDc * Sc;
  const int kend = qs + 16;

  for (int kb = 0; kb < kend; kb += 32) {
    v8f s0, s1;
#pragma unroll
    for (int j = 0; j < 8; ++j) { s0[j] = 0.f; s1[j] = 0.f; }

    const __bf16* kr0 = kbase + (size_t)(kb + n) * HDc;
    const __bf16* kr1 = kr0 + 16 * HDc;
#pragma unroll
    for (int c = 0; c < 4; ++c) {
      v16bf b0 = *(const v16bf*)(kr0 + c * 32 + half * 16);
      v16bf b1 = *(const v16bf*)(kr1 + c * 32 + half * 16);
      s0 = wmma_bf16(aq[c], b0, s0);
      s1 = wmma_bf16(aq[c], b1, s1);
    }

#pragma unroll
    for (int v = 0; v < 8; ++v) {
      const int qa = qs + v + 8 * half;
      float x0 = s0[v] * SCALE; if (kb + n      > qa) x0 = NEGBIG;
      float x1 = s1[v] * SCALE; if (kb + 16 + n > qa) x1 = NEGBIG;
      float rm = fmaxf(x0, x1);
      rm = fmaxf(rm, __shfl_xor(rm, 1));
      rm = fmaxf(rm, __shfl_xor(rm, 2));
      rm = fmaxf(rm, __shfl_xor(rm, 4));
      rm = fmaxf(rm, __shfl_xor(rm, 8));
      const float mn = fmaxf(mx[v], rm);
      const float al = __expf(mx[v] - mn);
      const float p0 = __expf(x0 - mn), p1 = __expf(x1 - mn);
      float rs = p0 + p1;
      rs += __shfl_xor(rs, 1); rs += __shfl_xor(rs, 2);
      rs += __shfl_xor(rs, 4); rs += __shfl_xor(rs, 8);
      ls[v] = ls[v] * al + rs;
      mx[v] = mn;
#pragma unroll
      for (int sl = 0; sl < 8; ++sl) o[sl][v] *= al;
      const int r = v + 8 * half;
      pbuf[wave][r * 32 + n]      = (__bf16)p0;   // C-layout -> row-major tile
      pbuf[wave][r * 32 + 16 + n] = (__bf16)p1;
    }
    // same-wave LDS transpose: DS ops are in-order; fence compiler + HW
    asm volatile("s_wait_dscnt 0" ::: "memory");

    v16bf ap = ldA(&pbuf[wave][n * 32 + half * 16],
                   &pbuf[wave][n * 32 + half * 16 + 8]);
#pragma unroll
    for (int sl = 0; sl < 8; ++sl) {
      const __bf16* vr = vbase + (size_t)(sl * 16 + n) * Sc + kb + half * 16;
      v16bf bv = *(const v16bf*)vr;
      o[sl] = wmma_bf16(ap, bv, o[sl]);
    }
  }

#pragma unroll
  for (int sl = 0; sl < 8; ++sl) {
#pragma unroll
    for (int v = 0; v < 8; ++v) {
      const int r = v + 8 * half;
      const float inv = 1.0f / ls[v];
      aws[(size_t)(b * Sc + qs + r) * Dc + h * HDc + sl * 16 + n] =
          (__bf16)(o[sl][v] * inv);
    }
  }
}

// ---------------------------------------------------------------------------
// Kernel 3: output projection  Y = attn_out @ Wo^T  (fp32 out).
//   grid = (B*S/128, D/128)   block = 256 (8 waves, 16 rows each)
// ---------------------------------------------------------------------------
__global__ __launch_bounds__(256) void oproj(
    const __bf16* __restrict__ aws, const __bf16* __restrict__ Wob,
    float* __restrict__ out) {
  __shared__ __attribute__((aligned(64))) __bf16 wtile[2][128 * 32];

  const int tid = threadIdx.x, wave = tid >> 5, lane = tid & 31;
  const int n = lane & 15, half = lane >> 4;
  const int rowbase = blockIdx.x * 128;
  const int colbase = blockIdx.y * 128;
  const __bf16* arow = aws + (size_t)(rowbase + wave * 16 + n) * Dc;

  v8f acc[8];
#pragma unroll
  for (int i = 0; i < 8; ++i)
#pragma unroll
    for (int j = 0; j < 8; ++j) acc[i][j] = 0.f;

  prefetch_wtile(Wob, colbase, 0, &wtile[0][0], tid);

  int it = 0;
  for (int kb = 0; kb < Dc; kb += 32, ++it) {
    const int nxt = kb + 32;
    if (nxt < Dc) {
      prefetch_wtile(Wob, colbase, nxt, &wtile[(it + 1) & 1][0], tid);
      asm volatile("s_wait_asynccnt 0x2" ::: "memory");
    } else {
      asm volatile("s_wait_asynccnt 0x0" ::: "memory");
    }
    __syncthreads();

    const __bf16* tile = &wtile[it & 1][0];
    v16bf a = ldA(arow + kb + half * 16, arow + kb + half * 16 + 8);
#pragma unroll
    for (int sl = 0; sl < 8; ++sl) {
      v16bf b = *(const v16bf*)&tile[(sl * 16 + n) * 32 + half * 16];
      acc[sl] = wmma_bf16(a, b, acc[sl]);
    }
    __syncthreads();
  }

#pragma unroll
  for (int sl = 0; sl < 8; ++sl) {
#pragma unroll
    for (int v = 0; v < 8; ++v) {
      const int r = v + 8 * half;
      out[(size_t)(rowbase + wave * 16 + r) * Dc + colbase + sl * 16 + n] =
          acc[sl][v];
    }
  }
}

// ---------------------------------------------------------------------------
extern "C" void kernel_launch(void* const* d_in, const int* in_sizes, int n_in,
                              void* d_out, int out_size, void* d_ws,
                              size_t ws_size, hipStream_t stream) {
  const float* hs = (const float*)d_in[0];
  const float* Wq = (const float*)d_in[1];
  const float* Wk = (const float*)d_in[2];
  const float* Wv = (const float*)d_in[3];
  const float* Wo = (const float*)d_in[4];
  // d_in[5] = position_ids (arange, implied by s index)

  const size_t nq = (size_t)Hc * HDc * Dc;   // 16.7M
  const size_t nk = (size_t)KVc * HDc * Dc;  // 4.2M
  const size_t nh = (size_t)Bc * Sc * Dc;    // 16.7M

  char* w = (char*)d_ws;
  __bf16* Wqb  = (__bf16*)w; w += nq * sizeof(__bf16);
  __bf16* Wkb  = (__bf16*)w; w += nk * sizeof(__bf16);
  __bf16* Wvb  = (__bf16*)w; w += nk * sizeof(__bf16);
  __bf16* Wob  = (__bf16*)w; w += nq * sizeof(__bf16);
  __bf16* hsb  = (__bf16*)w; w += nh * sizeof(__bf16);
  __bf16* qws  = (__bf16*)w; w += (size_t)Bc * Hc  * Sc * HDc * sizeof(__bf16);
  __bf16* kws  = (__bf16*)w; w += (size_t)Bc * KVc * Sc * HDc * sizeof(__bf16);
  __bf16* vtws = (__bf16*)w; w += (size_t)Bc * KVc * HDc * Sc * sizeof(__bf16);
  __bf16* aws  = (__bf16*)w;

  cvt_bf16<<<dim3(nq / (256 * 8)), 256, 0, stream>>>(Wq, Wqb, (int)nq);
  cvt_bf16<<<dim3(nk / (256 * 8)), 256, 0, stream>>>(Wk, Wkb, (int)nk);
  cvt_bf16<<<dim3(nk / (256 * 8)), 256, 0, stream>>>(Wv, Wvb, (int)nk);
  cvt_bf16<<<dim3(nq / (256 * 8)), 256, 0, stream>>>(Wo, Wob, (int)nq);
  cvt_bf16<<<dim3(nh / (256 * 8)), 256, 0, stream>>>(hs, hsb, (int)nh);

  qkv_rope<<<dim3((Bc * Sc) / 128, Hc + 2 * KVc), 256, 0, stream>>>(
      hsb, Wqb, Wkb, Wvb, qws, kws, vtws);
  attn<<<dim3((Bc * Hc * (Sc / 16)) / 8), 256, 0, stream>>>(qws, kws, vtws, aws);
  oproj<<<dim3((Bc * Sc) / 128, Dc / 128), 256, 0, stream>>>(aws, Wob,
                                                             (float*)d_out);
}
